// Generator_33397665694018
// MI455X (gfx1250) — matrix operand order repaired and checked by reading, not compiled, not run on hardware
//
#include <hip/hip_runtime.h>

#define V 32000
#define E 256
#define H 1024
#define B 128
#define T 32
#define NBLK (V / 16)  // 2000 column tiles of the vocab
#define NSUB 4         // N-tiles per workgroup in the packed logits kernel

#if __has_builtin(__builtin_amdgcn_global_load_async_to_lds_b128)
#define HAVE_ASYNC_LDS 1
#else
#define HAVE_ASYNC_LDS 0
#endif

typedef int v4i __attribute__((vector_size(16)));
typedef __attribute__((address_space(1))) v4i* gptr_v4i;
typedef __attribute__((address_space(3))) v4i* lptr_v4i;

__device__ __forceinline__ void wait_async_lds() {
#if HAVE_ASYNC_LDS
#if __has_builtin(__builtin_amdgcn_s_wait_asynccnt)
  __builtin_amdgcn_s_wait_asynccnt(0);
#else
  asm volatile("s_wait_asynccnt 0" ::: "memory");
#endif
#endif
}

typedef __attribute__((ext_vector_type(16))) _Float16 v16h;
typedef __attribute__((ext_vector_type(8)))  _Float16 v8h;
typedef __attribute__((ext_vector_type(8)))  float    v8f;

// ---------------------------------------------------------------------------
// fp32 -> (f16 hi, f16 lo) split helpers.  a ~= hi + lo with ~22 mantissa bits.
// ---------------------------------------------------------------------------
__device__ __forceinline__ void cvt8(const float* __restrict__ p, v16h& hi, v16h& lo, int e0) {
  const float4* p4 = (const float4*)p;   // callers guarantee 16B alignment
  float4 u = p4[0], v = p4[1];
  float f[8] = {u.x, u.y, u.z, u.w, v.x, v.y, v.z, v.w};
#pragma unroll
  for (int i = 0; i < 8; ++i) {
    _Float16 h = (_Float16)f[i];
    hi[e0 + i] = h;
    lo[e0 + i] = (_Float16)(f[i] - (float)h);
  }
}

__device__ __forceinline__ void cvt16c(const float* __restrict__ p, v16h& hi, v16h& lo) {
  cvt8(p, hi, lo, 0);
  cvt8(p + 8, hi, lo, 8);
}

// A-tile (16x32) fragment: lane half 0 holds K {kb..kb+15},
// lane half 1 holds K {kb+16..kb+31}.  'rowp' is this lane's M row pointer.
__device__ __forceinline__ void load_a_frag(const float* __restrict__ rowp, int kb, int koff,
                                            v16h& ah, v16h& al) {
  cvt8(rowp + kb + 2 * koff, ah, al, 0);
  cvt8(rowp + kb + 2 * koff + 8, ah, al, 8);
}

// Load 16 contiguous packed halfs (16B aligned) as a B fragment half.
__device__ __forceinline__ v16h load_h16(const _Float16* __restrict__ p) {
  v8h a = *(const v8h*)p;
  v8h b = *(const v8h*)(p + 8);
  v16h r;
#pragma unroll
  for (int i = 0; i < 8; ++i) { r[i] = a[i]; r[i + 8] = b[i]; }
  return r;
}

// D = A*B + C in ~fp32 via 3 f16 WMMAs (hi*hi + hi*lo + lo*hi).
__device__ __forceinline__ v8f wmma_split(v16h ah, v16h al, v16h bh, v16h bl, v8f c) {
  c = __builtin_amdgcn_wmma_f32_16x16x32_f16(false, ah, false, bh, (short)0, c, false, false);
  c = __builtin_amdgcn_wmma_f32_16x16x32_f16(false, ah, false, bl, (short)0, c, false, false);
  c = __builtin_amdgcn_wmma_f32_16x16x32_f16(false, al, false, bh, (short)0, c, false, false);
  return c;
}

__device__ __forceinline__ float sigmoidf_(float x) { return 1.0f / (1.0f + expf(-x)); }

// ---------------------------------------------------------------------------
// Weight pre-packing: fp32 -> f16 hi plane + f16 lo plane (once per launch,
// amortized over T=32 timesteps; ~12us of HBM traffic at 23.3 TB/s).
// ---------------------------------------------------------------------------
__global__ void pack_f16pair(const float* __restrict__ src, _Float16* __restrict__ hi,
                             _Float16* __restrict__ lo, int n) {
  const int i = blockIdx.x * blockDim.x + threadIdx.x;
  if (i < n) {
    const float f = src[i];
    const _Float16 h = (_Float16)f;
    hi[i] = h;
    lo[i] = (_Float16)(f - (float)h);
  }
}

// ---------------------------------------------------------------------------
// Kernel 1 (packed): fused embed + GRU cell, weights from f16 hi/lo planes.
// grid = (B/16, H/16), block = 32 (one wave) -> one 16x16 tile of h_new.
// ---------------------------------------------------------------------------
__global__ __launch_bounds__(32) void gru_gates_packed(
    const int* __restrict__ inp, int t,
    const float* __restrict__ emb,
    const _Float16* __restrict__ WihHi, const _Float16* __restrict__ WihLo,
    const _Float16* __restrict__ WhhHi, const _Float16* __restrict__ WhhLo,
    const float* __restrict__ bih, const float* __restrict__ bhh,
    const float* __restrict__ hin, float* __restrict__ hout) {
  const int lane = threadIdx.x;
  const int half = lane >> 4;
  const int l15  = lane & 15;
  const int koff = 8 * half;
  const int bm = blockIdx.x;
  const int nt = blockIdx.y;

  const int   arow = bm * 16 + l15;
  const int   tok  = inp[arow * T + t];
  const float* xrow = emb + (size_t)tok * E;
  const float* hrow = hin + (size_t)arow * H;

  const int jn = nt * 16 + l15;
  const size_t rr0 = (size_t)jn * E, rz0 = (size_t)(jn + H) * E, rn0 = (size_t)(jn + 2 * H) * E;
  const size_t sr0 = (size_t)jn * H, sz0 = (size_t)(jn + H) * H, sn0 = (size_t)(jn + 2 * H) * H;

  v8f air = {}, aiz = {}, ain = {}, ahr = {}, ahz = {}, ahn = {};

  for (int kb = 0; kb < E; kb += 32) {      // gi = x @ W_ih^T
    v16h ah, al; load_a_frag(xrow, kb, koff, ah, al);
    const int k16 = kb + 16 * half;
    air = wmma_split(ah, al, load_h16(WihHi + rr0 + k16), load_h16(WihLo + rr0 + k16), air);
    aiz = wmma_split(ah, al, load_h16(WihHi + rz0 + k16), load_h16(WihLo + rz0 + k16), aiz);
    ain = wmma_split(ah, al, load_h16(WihHi + rn0 + k16), load_h16(WihLo + rn0 + k16), ain);
  }
  for (int kb = 0; kb < H; kb += 32) {      // gh = h @ W_hh^T
    v16h ah, al; load_a_frag(hrow, kb, koff, ah, al);
    const int k16 = kb + 16 * half;
    ahr = wmma_split(ah, al, load_h16(WhhHi + sr0 + k16), load_h16(WhhLo + sr0 + k16), ahr);
    ahz = wmma_split(ah, al, load_h16(WhhHi + sz0 + k16), load_h16(WhhLo + sz0 + k16), ahz);
    ahn = wmma_split(ah, al, load_h16(WhhHi + sn0 + k16), load_h16(WhhLo + sn0 + k16), ahn);
  }

  const int j = nt * 16 + l15;
  const float bir = bih[j],         bhr = bhh[j];
  const float biz = bih[j + H],     bhz = bhh[j + H];
  const float bin = bih[j + 2 * H], bhn = bhh[j + 2 * H];
#pragma unroll
  for (int r = 0; r < 8; ++r) {
    const int b = bm * 16 + r + 8 * half;
    const float rg = sigmoidf_(air[r] + bir + ahr[r] + bhr);
    const float zg = sigmoidf_(aiz[r] + biz + ahz[r] + bhz);
    const float ng = tanhf(ain[r] + bin + rg * (ahn[r] + bhn));
    const float hp = hin[(size_t)b * H + j];
    hout[(size_t)b * H + j] = (1.0f - zg) * ng + zg * hp;
  }
}

// ---------------------------------------------------------------------------
// Kernel 2 (packed): logits GEMM fused with online-softmax partials.
// grid = NBLK/NSUB (64-col tiles), block = 256 (8 waves).  Wave w owns batch
// rows [16w,16w+16) and NSUB accumulator tiles.  The packed W_out tile is
// staged into double-buffered LDS via GLOBAL_LOAD_ASYNC_TO_LDS_B128
// (ASYNCcnt), so the next chunk's L2->LDS transfer overlaps the current
// chunk's 12 WMMAs, with one barrier per iteration.
// ---------------------------------------------------------------------------
#define KSTR_U 20  // uints per (column, 32-K chunk): 40 halfs, keeps 16B alignment

__global__ __launch_bounds__(256) void logits_softmax_packed(
    const float* __restrict__ hin,
    const _Float16* __restrict__ WoutHi, const _Float16* __restrict__ WoutLo,
    const float* __restrict__ bout,
    const int* __restrict__ target, int t,
    float* __restrict__ pmax, float* __restrict__ psum, float* __restrict__ tgtlog) {
  __shared__ unsigned int BsHi[2][64 * KSTR_U];
  __shared__ unsigned int BsLo[2][64 * KSTR_U];

  const int tid  = threadIdx.x;
  const int lane = tid & 31;
  const int w    = tid >> 5;
  const int half = lane >> 4;
  const int l15  = lane & 15;
  const int koff = 8 * half;
  const int blk  = blockIdx.x;          // 64-column tile
  const int nglob = blk * 64;

  const float* hrow = hin + (size_t)(w * 16 + l15) * H;

  // Stage 64 cols x 32 K of both planes (8 KB) = 512 x b128; 2 per thread.
  auto stage = [&](int kb, int buf) {
#pragma unroll
    for (int ii = 0; ii < 2; ++ii) {
      const int seg   = tid + ii * 256;     // 0..511
      const int plane = seg >> 8;
      const int col   = (seg >> 2) & 63;
      const int q     = seg & 3;            // which 16B quarter of the 32-K chunk
      const _Float16* gsrc =
          (plane ? WoutLo : WoutHi) + (size_t)(nglob + col) * H + kb + q * 8;
      unsigned int* ldst =
          (plane ? &BsLo[buf][0] : &BsHi[buf][0]) + col * KSTR_U + q * 4;
#if HAVE_ASYNC_LDS
      __builtin_amdgcn_global_load_async_to_lds_b128(
          (gptr_v4i)(gsrc), (lptr_v4i)(ldst), 0, 0);
#else
      *(uint4*)ldst = *(const uint4*)gsrc;
#endif
    }
  };

  v8f acc[NSUB] = {};
  const int NITER = H / 32;
  stage(0, 0);
  for (int it = 0; it < NITER; ++it) {
    const int kb  = it * 32;
    const int buf = it & 1;
    wait_async_lds();
    __syncthreads();                       // staged data visible to all 8 waves
    if (it + 1 < NITER) stage(kb + 32, buf ^ 1);  // overlaps with WMMAs below

    v16h ah, al; load_a_frag(hrow, kb, koff, ah, al);
#pragma unroll
    for (int s = 0; s < NSUB; ++s) {
      const int cs = s * 16 + l15;
      const _Float16* ph = (const _Float16*)&BsHi[buf][cs * KSTR_U] + 16 * half;
      const _Float16* pl = (const _Float16*)&BsLo[buf][cs * KSTR_U] + 16 * half;
      acc[s] = wmma_split(ah, al, load_h16(ph), load_h16(pl), acc[s]);
    }
    // No trailing barrier: buffer buf^1 was last read at it-1, proven drained
    // by this iteration's top barrier before its next overwrite at it+1.
  }

  // Epilogue per 16-col subtile: bias, target capture, row (max,sumexp) partials.
#pragma unroll
  for (int s = 0; s < NSUB; ++s) {
    const int colblk = blk * NSUB + s;
    const float bo = bout[colblk * 16 + l15];
    float vals[8];
#pragma unroll
    for (int r = 0; r < 8; ++r) vals[r] = acc[s][r] + bo;

#pragma unroll
    for (int r = 0; r < 8; ++r) {
      const int b  = w * 16 + r + 8 * half;
      const int tt = target[b * T + t];
      if ((tt >> 4) == colblk && (tt & 15) == l15) tgtlog[b] = vals[r];  // unique writer
    }
#pragma unroll
    for (int r = 0; r < 8; ++r) {
      float mv = vals[r];
#pragma unroll
      for (int mask = 1; mask < 16; mask <<= 1) mv = fmaxf(mv, __shfl_xor(mv, mask, 32));
      float sv = expf(vals[r] - mv);
#pragma unroll
      for (int mask = 1; mask < 16; mask <<= 1) sv += __shfl_xor(sv, mask, 32);
      if (l15 == 0) {
        const int b = w * 16 + r + 8 * half;
        pmax[(size_t)b * NBLK + colblk] = mv;
        psum[(size_t)b * NBLK + colblk] = sv;
      }
    }
  }
}

// ---------------------------------------------------------------------------
// Fallback kernels (convert weights on the fly) if ws_size can't hold planes.
// ---------------------------------------------------------------------------
__global__ __launch_bounds__(32) void gru_gates_wmma(
    const int* __restrict__ inp, int t, const float* __restrict__ emb,
    const float* __restrict__ Wih, const float* __restrict__ Whh,
    const float* __restrict__ bih, const float* __restrict__ bhh,
    const float* __restrict__ hin, float* __restrict__ hout) {
  const int lane = threadIdx.x, half = lane >> 4, l15 = lane & 15, koff = 8 * half;
  const int bm = blockIdx.x, nt = blockIdx.y;
  const int arow = bm * 16 + l15;
  const int tok = inp[arow * T + t];
  const float* xrow = emb + (size_t)tok * E;
  const float* hrow = hin + (size_t)arow * H;
  const int jn = nt * 16 + l15;
  const float* wr = Wih + (size_t)jn * E;
  const float* wz = Wih + (size_t)(jn + H) * E;
  const float* wn = Wih + (size_t)(jn + 2 * H) * E;
  const float* vr = Whh + (size_t)jn * H;
  const float* vz = Whh + (size_t)(jn + H) * H;
  const float* vn = Whh + (size_t)(jn + 2 * H) * H;
  v8f air = {}, aiz = {}, ain = {}, ahr = {}, ahz = {}, ahn = {};
  for (int kb = 0; kb < E; kb += 32) {
    v16h ah, al; load_a_frag(xrow, kb, koff, ah, al);
    const int k16 = kb + 16 * half;
    v16h bh, bl;
    cvt16c(wr + k16, bh, bl); air = wmma_split(ah, al, bh, bl, air);
    cvt16c(wz + k16, bh, bl); aiz = wmma_split(ah, al, bh, bl, aiz);
    cvt16c(wn + k16, bh, bl); ain = wmma_split(ah, al, bh, bl, ain);
  }
  for (int kb = 0; kb < H; kb += 32) {
    v16h ah, al; load_a_frag(hrow, kb, koff, ah, al);
    const int k16 = kb + 16 * half;
    v16h bh, bl;
    cvt16c(vr + k16, bh, bl); ahr = wmma_split(ah, al, bh, bl, ahr);
    cvt16c(vz + k16, bh, bl); ahz = wmma_split(ah, al, bh, bl, ahz);
    cvt16c(vn + k16, bh, bl); ahn = wmma_split(ah, al, bh, bl, ahn);
  }
  const int j = nt * 16 + l15;
  const float bir = bih[j], bhr = bhh[j];
  const float biz = bih[j + H], bhz = bhh[j + H];
  const float bin = bih[j + 2 * H], bhn = bhh[j + 2 * H];
#pragma unroll
  for (int r = 0; r < 8; ++r) {
    const int b = bm * 16 + r + 8 * half;
    const float rg = sigmoidf_(air[r] + bir + ahr[r] + bhr);
    const float zg = sigmoidf_(aiz[r] + biz + ahz[r] + bhz);
    const float ng = tanhf(ain[r] + bin + rg * (ahn[r] + bhn));
    const float hp = hin[(size_t)b * H + j];
    hout[(size_t)b * H + j] = (1.0f - zg) * ng + zg * hp;
  }
}

#define BS_STRIDE 36
__global__ __launch_bounds__(256) void logits_softmax_wmma(
    const float* __restrict__ hin, const float* __restrict__ Wout,
    const float* __restrict__ bout, const int* __restrict__ target, int t,
    float* __restrict__ pmax, float* __restrict__ psum, float* __restrict__ tgtlog) {
  __shared__ float Bs[16 * BS_STRIDE];
  const int tid = threadIdx.x, lane = tid & 31, w = tid >> 5;
  const int half = lane >> 4, l15 = lane & 15, koff = 8 * half;
  const int blk = blockIdx.x, nglob = blk * 16;
  const float* hrow = hin + (size_t)(w * 16 + l15) * H;
  v8f acc = {};
  for (int kb = 0; kb < H; kb += 32) {
    for (int i = tid; i < 512; i += 256) {
      const int nl = i >> 5, kk = i & 31;
      Bs[nl * BS_STRIDE + kk] = Wout[(size_t)(nglob + nl) * H + kb + kk];
    }
    __syncthreads();
    v16h ah, al; load_a_frag(hrow, kb, koff, ah, al);
    v16h bh, bl; cvt16c(&Bs[l15 * BS_STRIDE + 16 * half], bh, bl);
    acc = wmma_split(ah, al, bh, bl, acc);
    __syncthreads();
  }
  const float bo = bout[nglob + l15];
  float vals[8];
#pragma unroll
  for (int r = 0; r < 8; ++r) vals[r] = acc[r] + bo;
#pragma unroll
  for (int r = 0; r < 8; ++r) {
    const int b = w * 16 + r + 8 * half;
    const int tt = target[b * T + t];
    if ((tt >> 4) == blk && (tt & 15) == l15) tgtlog[b] = vals[r];
  }
#pragma unroll
  for (int r = 0; r < 8; ++r) {
    float mv = vals[r];
#pragma unroll
    for (int mask = 1; mask < 16; mask <<= 1) mv = fmaxf(mv, __shfl_xor(mv, mask, 32));
    float sv = expf(vals[r] - mv);
#pragma unroll
    for (int mask = 1; mask < 16; mask <<= 1) sv += __shfl_xor(sv, mask, 32);
    if (l15 == 0) {
      const int b = w * 16 + r + 8 * half;
      pmax[(size_t)b * NBLK + blk] = mv;
      psum[(size_t)b * NBLK + blk] = sv;
    }
  }
}

// ---------------------------------------------------------------------------
// Kernel 3: merge 2000 softmax partials per row -> NLL, accumulate per row.
// ---------------------------------------------------------------------------
__global__ __launch_bounds__(256) void softmax_reduce(
    const float* __restrict__ pmax, const float* __restrict__ psum,
    const float* __restrict__ tgtlog, float* __restrict__ lacc) {
  const int b = blockIdx.x, tid = threadIdx.x;
  float m = -INFINITY, s = 0.0f;
  for (int i = tid; i < NBLK; i += 256) {
    const float pm = pmax[(size_t)b * NBLK + i];
    const float ps = psum[(size_t)b * NBLK + i];
    const float nm = fmaxf(m, pm);
    s = s * expf(m - nm) + ps * expf(pm - nm);
    m = nm;
  }
  __shared__ float sm[256], ss[256];
  sm[tid] = m; ss[tid] = s;
  __syncthreads();
  for (int o = 128; o > 0; o >>= 1) {
    if (tid < o) {
      const float m2 = sm[tid + o], s2 = ss[tid + o];
      const float nm = fmaxf(sm[tid], m2);
      ss[tid] = ss[tid] * expf(sm[tid] - nm) + s2 * expf(m2 - nm);
      sm[tid] = nm;
    }
    __syncthreads();
  }
  if (tid == 0) lacc[b] += -(tgtlog[b] - sm[0] - logf(ss[0]));
}

__global__ __launch_bounds__(128) void finalize(const float* __restrict__ lacc,
                                                float* __restrict__ out) {
  __shared__ float sm[128];
  sm[threadIdx.x] = lacc[threadIdx.x];
  __syncthreads();
  for (int o = 64; o > 0; o >>= 1) {
    if (threadIdx.x < o) sm[threadIdx.x] += sm[threadIdx.x + o];
    __syncthreads();
  }
  if (threadIdx.x == 0) out[0] = sm[0] / (float)B;
}

__global__ void zero_f32(float* __restrict__ p, int n) {
  const int i = blockIdx.x * blockDim.x + threadIdx.x;
  if (i < n) p[i] = 0.0f;
}

// ---------------------------------------------------------------------------
extern "C" void kernel_launch(void* const* d_in, const int* in_sizes, int n_in,
                              void* d_out, int out_size, void* d_ws, size_t ws_size,
                              hipStream_t stream) {
  const int*   inp  = (const int*)  d_in[0];
  const int*   tgt  = (const int*)  d_in[1];
  const float* emb  = (const float*)d_in[2];
  const float* Wih  = (const float*)d_in[3];
  const float* Whh  = (const float*)d_in[4];
  const float* bih  = (const float*)d_in[5];
  const float* bhh  = (const float*)d_in[6];
  const float* Wout = (const float*)d_in[7];
  const float* bout = (const float*)d_in[8];
  float* out = (float*)d_out;

  // Workspace carve-out (256B aligned blocks).
  char*  wsp = (char*)d_ws;
  size_t cur = 0;
  auto alloc = [&](size_t bytes) {
    char* p = wsp + cur;
    cur += (bytes + 255) & ~(size_t)255;
    return p;
  };
  float* h0   = (float*)alloc((size_t)B * H * 4);
  float* h1   = (float*)alloc((size_t)B * H * 4);
  float* pmax = (float*)alloc((size_t)B * NBLK * 4);
  float* psum = (float*)alloc((size_t)B * NBLK * 4);
  float* tlg  = (float*)alloc((size_t)B * 4);
  float* lacc = (float*)alloc((size_t)B * 4);
  // Packed f16 hi/lo weight planes (~140 MB; W_out hi+lo == same bytes as fp32,
  // so the L2-residency argument (192 MB) is unchanged).
  _Float16* WoutHi = (_Float16*)alloc((size_t)V * H * 2);
  _Float16* WoutLo = (_Float16*)alloc((size_t)V * H * 2);
  _Float16* WihHi  = (_Float16*)alloc((size_t)3 * H * E * 2);
  _Float16* WihLo  = (_Float16*)alloc((size_t)3 * H * E * 2);
  _Float16* WhhHi  = (_Float16*)alloc((size_t)3 * H * H * 2);
  _Float16* WhhLo  = (_Float16*)alloc((size_t)3 * H * H * 2);
  const bool packed_ok = (cur <= ws_size);

  zero_f32<<<dim3((B * H + 255) / 256), 256, 0, stream>>>(h0, B * H);
  zero_f32<<<1, 128, 0, stream>>>(lacc, B);

  float* hc = h0;
  float* hn = h1;
  if (packed_ok) {
    const int nWout = V * H, nWih = 3 * H * E, nWhh = 3 * H * H;
    pack_f16pair<<<dim3((nWout + 255) / 256), 256, 0, stream>>>(Wout, WoutHi, WoutLo, nWout);
    pack_f16pair<<<dim3((nWih  + 255) / 256), 256, 0, stream>>>(Wih,  WihHi,  WihLo,  nWih);
    pack_f16pair<<<dim3((nWhh  + 255) / 256), 256, 0, stream>>>(Whh,  WhhHi,  WhhLo,  nWhh);
    for (int t = 0; t < T; ++t) {
      gru_gates_packed<<<dim3(B / 16, H / 16), 32, 0, stream>>>(
          inp, t, emb, WihHi, WihLo, WhhHi, WhhLo, bih, bhh, hc, hn);
      logits_softmax_packed<<<NBLK / NSUB, 256, 0, stream>>>(
          hn, WoutHi, WoutLo, bout, tgt, t, pmax, psum, tlg);
      softmax_reduce<<<B, 256, 0, stream>>>(pmax, psum, tlg, lacc);
      float* tmp = hc; hc = hn; hn = tmp;
    }
  } else {
    for (int t = 0; t < T; ++t) {
      gru_gates_wmma<<<dim3(B / 16, H / 16), 32, 0, stream>>>(
          inp, t, emb, Wih, Whh, bih, bhh, hc, hn);
      logits_softmax_wmma<<<NBLK, 256, 0, stream>>>(
          hn, Wout, bout, tgt, t, pmax, psum, tlg);
      softmax_reduce<<<B, 256, 0, stream>>>(pmax, psum, tlg, lacc);
      float* tmp = hc; hc = hn; hn = tmp;
    }
  }
  finalize<<<1, B, 0, stream>>>(lacc, out);

  (void)in_sizes; (void)n_in; (void)out_size;
}
